// mymodel_81758997446943
// MI455X (gfx1250) — matrix operand order repaired and checked by reading, not compiled, not run on hardware
//
#include <hip/hip_runtime.h>
#include <hip/hip_bf16.h>

// ---------------------------------------------------------------------------
// Types for CDNA5 WMMA fragments (wave32): A/B = 16 halves/lane, C/D = 8 f32.
// ---------------------------------------------------------------------------
typedef __attribute__((ext_vector_type(16))) _Float16 v16h;
typedef __attribute__((ext_vector_type(8)))  _Float16 v8h;
typedef __attribute__((ext_vector_type(4)))  _Float16 v4h;
typedef __attribute__((ext_vector_type(8)))  float    v8f;

#define SEQ   512
#define BATCHN 256
#define EMBD  128
#define HID   64
#define GATES 256          // 4*HID
#define HPAD  72           // padded row stride (halves) for 64-wide f16 rows
#define H32P  68           // padded row stride (floats) for h32 buffer

__device__ __forceinline__ float sigf(float x) { return 1.0f / (1.0f + __expf(-x)); }

__device__ __forceinline__ v8f wmma_f16(v16h a, v16h b, v8f c) {
    // D = A(16x32 f16) * B(32x16 f16) + C(16x16 f32)
    return __builtin_amdgcn_wmma_f32_16x16x32_f16(
        /*neg_a=*/false, a, /*neg_b=*/false, b,
        /*c_mod=*/(short)0, c, /*reuse_a=*/false, /*reuse_b=*/false);
}

// ---------------------------------------------------------------------------
// Kernel 1: embedding gather + fp32 -> f16 convert.  out: (T, B, EMB) f16
// ---------------------------------------------------------------------------
__global__ __launch_bounds__(256)
void embed_kernel(const long long* __restrict__ idx,
                  const float* __restrict__ emb,
                  _Float16* __restrict__ out) {
    size_t e  = ((size_t)blockIdx.x * blockDim.x + threadIdx.x) * 4;
    size_t tb = e >> 7;                 // / EMB
    int    k  = (int)(e & (EMBD - 1));
    long long row = idx[tb];
    const float4 v = *(const float4*)(emb + (size_t)row * EMBD + k);
    v4h h;
    h.x = (_Float16)v.x; h.y = (_Float16)v.y;
    h.z = (_Float16)v.z; h.w = (_Float16)v.w;
    *(v4h*)(out + e) = h;
}

// ---------------------------------------------------------------------------
// Kernel 2: one LSTM layer for one 16-row batch strip per workgroup.
// One wave (32 lanes) per workgroup: all 16 N-tiles of the (16 x 256) gate
// matrix are accumulated by this wave; the four gates of each hidden unit
// land in the same lane, so gating is done fully in registers.
// ---------------------------------------------------------------------------
__global__ __launch_bounds__(32)
void lstm_strip_kernel(const _Float16* __restrict__ xin, int D,
                       const float* __restrict__ Wih,
                       const float* __restrict__ Whh,
                       const float* __restrict__ bih,
                       const float* __restrict__ bhh,
                       _Float16* __restrict__ hout,   // (T,B,H) f16, next layer in
                       float* __restrict__ sm_out,    // final softmax out (fp32)
                       int is_final) {
    extern __shared__ char smem[];
    const int Dp = D + 8;                               // padded W_ih row (halves)
    _Float16* sWih  = (_Float16*)smem;                  // GATES * Dp
    _Float16* sWhh  = sWih + GATES * Dp;                // GATES * HPAD
    float*    sBias = (float*)(sWhh + GATES * HPAD);    // GATES
    float*    sH32  = sBias + GATES;                    // 16 * H32P (final layer)
    _Float16* sHf   = (_Float16*)(sH32 + 16 * H32P);    // 16 * HPAD  (h state f16)

    const int tid = threadIdx.x;                        // == lane (wave32)
    const int b0  = blockIdx.x * 16;                    // batch strip base

    // ---- one-time preamble: weights/bias -> LDS (f16), zero h -------------
    for (int i = tid; i < GATES * D; i += 32) {
        int r = i / D, k = i - r * D;
        sWih[r * Dp + k] = (_Float16)Wih[i];
    }
    for (int i = tid; i < GATES * HID; i += 32) {
        int r = i >> 6, k = i & 63;
        sWhh[r * HPAD + k] = (_Float16)Whh[i];
    }
    for (int i = tid; i < GATES; i += 32) sBias[i] = bih[i] + bhh[i];
    for (int i = tid; i < 16 * HPAD; i += 32) sHf[i] = (_Float16)0.0f;
    __syncthreads();

    // ---- per-lane fragment geometry ---------------------------------------
    const int p    = tid & 15;            // column-in-tile (N%16), B-frag row sel
    const int hi   = tid >> 4;            // 0 / 1 lane half
    const int arow = p;                   // A-frag row (M)
    const int koff = hi << 3;             // A-frag K sub-offset (halves): 0 or 8
    const int bko  = hi << 3;             // B-frag K offset (halves): 0 or 8
    const int mhi  = hi << 3;             // C-frag row base: 0 or 8
    const int KBX  = D >> 5;              // K blocks of 32 for the input GEMM

    float bias_c[16];
#pragma unroll
    for (int tn = 0; tn < 16; ++tn) bias_c[tn] = sBias[tn * 16 + p];

    float cst[8][4] = {};                 // c-state: 8 rows x 4 j-slots per lane

    for (int t = 0; t < SEQ; ++t) {
        const _Float16* xb = xin + ((size_t)t * BATCHN + b0 + arow) * D;
        // prefetch next timestep's x rows (global_prefetch_b8)
        __builtin_prefetch(xb + (size_t)BATCHN * D, 0, 1);

        v8f acc[16];
#pragma unroll
        for (int tn = 0; tn < 16; ++tn)
#pragma unroll
            for (int r = 0; r < 8; ++r) acc[tn][r] = bias_c[tn];

        // ---- x_t @ W_ih^T  (A from global, B resident in LDS) -------------
        for (int kb = 0; kb < KBX; ++kb) {
            union { v16h v; v8h h[2]; } a;
            const _Float16* ap = xb + kb * 32 + koff;
            a.h[0] = *(const v8h*)ap;
            a.h[1] = *(const v8h*)(ap + 16);
#pragma unroll
            for (int tn = 0; tn < 16; ++tn) {
                union { v16h v; v8h h[2]; } b;
                const _Float16* bp = sWih + (tn * 16 + p) * Dp + kb * 32 + bko;
                b.h[0] = *(const v8h*)bp;
                b.h[1] = *(const v8h*)(bp + 16);
                acc[tn] = wmma_f16(a.v, b.v, acc[tn]);
            }
        }
        // ---- h_{t-1} @ W_hh^T  (A and B both in LDS) -----------------------
#pragma unroll
        for (int kb = 0; kb < 2; ++kb) {
            union { v16h v; v8h h[2]; } a;
            const _Float16* ap = sHf + arow * HPAD + kb * 32 + koff;
            a.h[0] = *(const v8h*)ap;
            a.h[1] = *(const v8h*)(ap + 16);
#pragma unroll
            for (int tn = 0; tn < 16; ++tn) {
                union { v16h v; v8h h[2]; } b;
                const _Float16* bp = sWhh + (tn * 16 + p) * HPAD + kb * 32 + bko;
                b.h[0] = *(const v8h*)bp;
                b.h[1] = *(const v8h*)(bp + 16);
                acc[tn] = wmma_f16(a.v, b.v, acc[tn]);
            }
        }

        // ---- gating: i,f,g,o for unit j=16q+p are tiles q,4+q,8+q,12+q ----
        // (same lane, pure register math; only h needs the LDS transpose)
#pragma unroll
        for (int r = 0; r < 8; ++r) {
#pragma unroll
            for (int q = 0; q < 4; ++q) {
                float gi = acc[q][r];
                float gf = acc[4 + q][r];
                float gg = acc[8 + q][r];
                float go = acc[12 + q][r];
                float c  = sigf(gf) * cst[r][q] + sigf(gi) * tanhf(gg);
                cst[r][q] = c;
                float hv = sigf(go) * tanhf(c);
                int j = (q << 4) + p;
                int M = mhi + r;
                sHf[M * HPAD + j] = (_Float16)hv;     // next-step A fragment
                if (is_final) {
                    sH32[M * H32P + j] = hv;
                } else {
                    hout[((size_t)t * BATCHN + b0 + M) * HID + j] = (_Float16)hv;
                }
            }
        }
        __syncthreads();   // single-wave WG: cheap; orders sHf/sH32 for reuse

        if (is_final && tid < 16) {       // fused row softmax over HID=64
            float m = -1e30f;
            for (int j = 0; j < HID; ++j) m = fmaxf(m, sH32[tid * H32P + j]);
            float s = 0.0f;
            for (int j = 0; j < HID; ++j) s += __expf(sH32[tid * H32P + j] - m);
            float inv = 1.0f / s;
            float* op = sm_out + ((size_t)t * BATCHN + b0 + tid) * HID;
            for (int j = 0; j < HID; ++j)
                op[j] = __expf(sH32[tid * H32P + j] - m) * inv;
        }
        __syncthreads();
    }
}

// ---------------------------------------------------------------------------
static size_t lstm_smem_bytes(int D) {
    size_t b = (size_t)(GATES * (D + 8) + GATES * HPAD) * sizeof(_Float16);
    b += (size_t)GATES * sizeof(float);          // bias
    b += (size_t)16 * H32P * sizeof(float);      // h32
    b += (size_t)16 * HPAD * sizeof(_Float16);   // h f16
    return b;
}

extern "C" void kernel_launch(void* const* d_in, const int* in_sizes, int n_in,
                              void* d_out, int out_size, void* d_ws, size_t ws_size,
                              hipStream_t stream) {
    (void)in_sizes; (void)n_in; (void)out_size; (void)ws_size;

    const long long* x    = (const long long*)d_in[0];
    const float*     emb  = (const float*)d_in[1];
    const float*     Wih1 = (const float*)d_in[2];
    const float*     Whh1 = (const float*)d_in[3];
    const float*     bih1 = (const float*)d_in[4];
    const float*     bhh1 = (const float*)d_in[5];
    const float*     Wih2 = (const float*)d_in[6];
    const float*     Whh2 = (const float*)d_in[7];
    const float*     bih2 = (const float*)d_in[8];
    const float*     bhh2 = (const float*)d_in[9];
    const float*     Wih3 = (const float*)d_in[10];
    const float*     Whh3 = (const float*)d_in[11];
    const float*     bih3 = (const float*)d_in[12];
    const float*     bhh3 = (const float*)d_in[13];
    float* out = (float*)d_out;

    // workspace: ping-pong f16 activation buffers (32 MB + 16 MB)
    _Float16* bufA = (_Float16*)d_ws;                         // (T,B,EMB) f16
    _Float16* bufB = bufA + (size_t)SEQ * BATCHN * EMBD;      // (T,B,HID) f16

    (void)hipFuncSetAttribute((const void*)lstm_strip_kernel,
                              hipFuncAttributeMaxDynamicSharedMemorySize, 131072);

    // 1) embedding gather + f16 convert
    {
        int total4 = SEQ * BATCHN * EMBD / 4;
        embed_kernel<<<total4 / 256, 256, 0, stream>>>(x, emb, bufA);
    }

    size_t s1 = lstm_smem_bytes(EMBD);
    size_t s2 = lstm_smem_bytes(HID);

    // 2) three stacked LSTM layers; 16 independent batch strips each
    lstm_strip_kernel<<<BATCHN / 16, 32, s1, stream>>>(
        bufA, EMBD, Wih1, Whh1, bih1, bhh1, bufB, nullptr, 0);
    lstm_strip_kernel<<<BATCHN / 16, 32, s2, stream>>>(
        bufB, HID, Wih2, Whh2, bih2, bhh2, bufA, nullptr, 0);
    lstm_strip_kernel<<<BATCHN / 16, 32, s2, stream>>>(
        bufA, HID, Wih3, Whh3, bih3, bhh3, nullptr, out, 1);
}
